// MultiHeadAttention_66005057405547
// MI455X (gfx1250) — matrix operand order repaired and checked by reading, not compiled, not run on hardware
//
#include <hip/hip_runtime.h>
#include <hip/hip_bf16.h>

// ---------------------------------------------------------------------------
// Graph-attention MHA for MI455X (gfx1250, wave32, WMMA).
// Dominant cost: scores = ReLU(Q Kᵀ) @ rescale(dist)  ==  8 × 4096³ MACs
// (~1.1 TFLOP) -> heavily compute-bound -> bf16 WMMA (16x16x32, f32 accum).
// All large intermediates are bf16 and sized to stay L2-resident (192 MB):
//   rescale(dist)ᵀ = 32 MB, per-head ReLU(QKᵀ) = 32 MB (heads sequential).
// attn_ctx uses 32-row tiles so every B-fragment feeds 2 WMMAs, halving the
// dominant L2 read stream vs a 16-row tiling.  Softmax is 2-pass (max,
// exp+sum) with b128 LDS accesses; 1/rowsum is folded into the final write.
//
// Workspace layout (requires ws_size >= 70 MiB):
//   qb  : bf16 [8][4096][32]   (2 MiB)
//   kb  : bf16 [8][4096][32]   (2 MiB)
//   vbT : bf16 [8][32][4096]   (2 MiB)  V transposed per head
//   dpT : bf16 [4096][4096]    (32 MiB) rescale(dist) TRANSPOSED
//   ab  : bf16 [4096][4096]    (32 MiB) ReLU(Q Kᵀ) for current head (reused)
// ---------------------------------------------------------------------------

typedef __attribute__((ext_vector_type(16))) __bf16 v16bf;
typedef __attribute__((ext_vector_type(8)))  float  v8f;

#define NN   4096
#define DM   256
#define NH   8
#define DH   32
#define ROWS 32    // rows per attn_ctx workgroup (2 WMMA row-tiles)
#define SPAD 4112  // padded LDS row stride (bf16 elems) for the score tile

union BF16x16 { v16bf v; uint4 q[2]; };
union BF16x8  { uint4 q; __bf16 e[8]; };

// A-fragment (16x32 bf16): lane<16 holds row M=lane, k={0..7,16..23}+base;
// lane>=16 holds row M=lane-16, k={8..15,24..31}. p points at (row, k0+base0).
__device__ inline v16bf load_a_frag(const __bf16* p) {
  BF16x16 r;
  r.q[0] = *(const uint4*)(p);        // 8 bf16: k0+base0 .. +7
  r.q[1] = *(const uint4*)(p + 16);   // 8 bf16: k0+base0+16 .. +23
  return r.v;
}

// B-fragment (32x16 bf16): lane<16 holds col N=lane, k={0..7,16..23};
// lane>=16 holds col N=lane-16, k={8..15,24..31}. p points at (col, k0+base0).
__device__ inline v16bf load_b_frag(const __bf16* p) {
  BF16x16 r;
  r.q[0] = *(const uint4*)(p);
  r.q[1] = *(const uint4*)(p + 16);
  return r.v;
}

#define WMMA_BF16(A, B, C) \
  __builtin_amdgcn_wmma_f32_16x16x32_bf16(false, (A), false, (B), (short)0, (C), false, false)

// ---------------------------------------------------------------------------
// Kernel 1: one projection  P = X @ Wᵀ  (4096x256 · 256x256), bf16 out.
// transposed==0: dst[h][n][d] (Q,K).  transposed==1: dst[h][d][n] (V).
// grid (32, 16), block 256 (8 waves; one 16x16 output tile per wave).
__global__ __launch_bounds__(256) void proj_one(
    const float* __restrict__ X, const float* __restrict__ W,
    __bf16* __restrict__ dst, int transposed) {
  const int wid  = threadIdx.x >> 5;
  const int lane = threadIdx.x & 31;
  const int l15  = lane & 15;
  const int hi   = lane >> 4;
  const int base0 = hi * 8, koff = hi * 16, rowoff = hi * 8;

  const int n0 = (blockIdx.x * 8 + wid) * 16;
  const int c0 = blockIdx.y * 16;            // output column tile (within 256)
  const int row = n0 + l15;
  const int col = c0 + l15;

  v8f acc = {};
  for (int kc = 0; kc < DM; kc += 32) {
    v16bf a, b;
#pragma unroll
    for (int i = 0; i < 8; ++i) {
      a[i]     = (__bf16)X[(size_t)row * DM + kc + base0 + i];
      a[i + 8] = (__bf16)X[(size_t)row * DM + kc + 16 + base0 + i];
    }
    // B(kk, jj) = W[c0+jj, kc+kk]  (X·Wᵀ): contiguous along k in W's row.
#pragma unroll
    for (int i = 0; i < 8; ++i) {
      b[i]     = (__bf16)W[(size_t)col * DM + kc + base0 + i];
      b[i + 8] = (__bf16)W[(size_t)col * DM + kc + 16 + base0 + i];
    }
    acc = WMMA_BF16(a, b, acc);
  }

  const int h = c0 >> 5;                     // 16-wide tile never crosses a head
  const int j = (c0 + l15) & 31;
  if (!transposed) {
#pragma unroll
    for (int r = 0; r < 8; ++r)
      dst[(size_t)h * NN * DH + (size_t)(n0 + rowoff + r) * DH + j] = (__bf16)acc[r];
  } else {
#pragma unroll
    for (int r = 0; r < 8; ++r)
      dst[(size_t)h * DH * NN + (size_t)j * NN + (n0 + rowoff + r)] = (__bf16)acc[r];
  }
}

// ---------------------------------------------------------------------------
// Kernel 2: dpT[p][m] = (1+e)/(1+exp(1-dist[m][p])) as bf16 (transposed store
// through an LDS tile so both the read and the write are coalesced).
// grid (128, 128), block (32, 8).
__global__ __launch_bounds__(256) void rescale_t(
    const float* __restrict__ dist, __bf16* __restrict__ dpT) {
  __shared__ float t[32][33];
  const int p0 = blockIdx.x * 32, m0 = blockIdx.y * 32;
#pragma unroll
  for (int s = 0; s < 4; ++s) {
    const int m = m0 + threadIdx.y + s * 8;
    const float d = dist[(size_t)m * NN + p0 + threadIdx.x];
    t[threadIdx.x][threadIdx.y + s * 8] = 3.7182818284f / (1.0f + __expf(1.0f - d));
  }
  __syncthreads();
#pragma unroll
  for (int s = 0; s < 4; ++s) {
    const int p = p0 + threadIdx.y + s * 8;
    dpT[(size_t)p * NN + m0 + threadIdx.x] = (__bf16)t[threadIdx.y + s * 8][threadIdx.x];
  }
}

// ---------------------------------------------------------------------------
// Kernel 3 (per head): ab = ReLU(Q_h K_hᵀ) bf16 [4096][4096].
// K-dim = dh = 32 -> exactly one WMMA per 16x16 tile. A-fragment (16 rows of
// Q_h) is loaded once per wave and reused across 16 m-tiles.
// grid 512, block 256 (8 waves; wave = one n-tile × 16 m-tiles).
__global__ __launch_bounds__(256) void qk_relu(
    const __bf16* __restrict__ qb, const __bf16* __restrict__ kb,
    __bf16* __restrict__ ab, int h) {
  const int wid  = threadIdx.x >> 5;
  const int lane = threadIdx.x & 31;
  const int l15  = lane & 15;
  const int hi   = lane >> 4;
  const int base0 = hi * 8, koff = hi * 16, rowoff = hi * 8;

  const int wg      = blockIdx.x * 8 + wid;   // 0..4095
  const int n0      = (wg >> 4) * 16;
  const int m_chunk = wg & 15;

  const __bf16* Q = qb + (size_t)h * NN * DH;
  const __bf16* K = kb + (size_t)h * NN * DH;

  const v16bf a = load_a_frag(Q + (size_t)(n0 + l15) * DH + base0);

  for (int t = 0; t < 16; ++t) {
    const int m0 = (m_chunk * 16 + t) * 16;
    const v16bf b = load_b_frag(K + (size_t)(m0 + l15) * DH + base0);
    v8f acc = {};
    acc = WMMA_BF16(a, b, acc);
#pragma unroll
    for (int r = 0; r < 8; ++r) {
      float v = acc[r];
      v = v > 0.0f ? v : 0.0f;                           // fused ReLU
      ab[(size_t)(n0 + rowoff + r) * NN + m0 + l15] = (__bf16)v;
    }
  }
}

// ---------------------------------------------------------------------------
// Kernel 4 (per head): fused  scores = ab @ dpT' /sqrt(32) -> softmax ->
// context = P @ V_h.  One workgroup (8 waves) per 32-row tile; each wave owns
// a 512-column strip processed in 4 passes of 8 col-tiles with TWO row-tiles
// live (acc[2][8] = 128 VGPRs) so every B-fragment feeds 2 WMMAs.
// Softmax: 2 passes (b128 LDS accesses); context accumulates the
// UNNORMALIZED exp() matrix and 1/rowsum is applied at the final store.
// Dynamic LDS: 32 x SPAD bf16 score tile (257 KB) + ctx + reduction scratch.
// grid 128, block 256.
__global__ __launch_bounds__(256) void attn_ctx(
    const __bf16* __restrict__ ab, const __bf16* __restrict__ dpT,
    const __bf16* __restrict__ vbT, float* __restrict__ out, int h) {
  extern __shared__ char smem[];
  __bf16* S       = (__bf16*)smem;                           // ROWS * SPAD bf16
  float*  ctx     = (float*)(smem + (size_t)ROWS * SPAD * 2); // 1024 f32
  float*  red     = ctx + ROWS * DH;                          // 256 f32
  float*  rowstat = red + 256;                                // 32 f32

  const int tid  = threadIdx.x;
  const int wid  = tid >> 5;
  const int lane = tid & 31;
  const int l15  = lane & 15;
  const int hi   = lane >> 4;
  const int base0 = hi * 8, koff = hi * 16, rowoff = hi * 8;
  const int n0 = blockIdx.x * ROWS;

  ctx[tid] = 0.0f; ctx[tid + 256] = 0.0f;
  ctx[tid + 512] = 0.0f; ctx[tid + 768] = 0.0f;

  // ---- stage A: score strip [32 x 512] per wave, K = 4096 -----------------
  const __bf16* Arow0 = ab + (size_t)(n0 + l15) * NN;        // row tile 0
  const __bf16* Arow1 = ab + (size_t)(n0 + 16 + l15) * NN;   // row tile 1
  const float scale = 0.17677669529663687f;                  // 1/sqrt(32)
  for (int pass = 0; pass < 4; ++pass) {
    const int cbase = wid * 512 + pass * 128;
    v8f acc[2][8];
    const v8f z = {};
#pragma unroll
    for (int t = 0; t < 8; ++t) { acc[0][t] = z; acc[1][t] = z; }

    for (int kc = 0; kc < NN; kc += 32) {
      const v16bf a0 = load_a_frag(Arow0 + kc + base0);
      const v16bf a1 = load_a_frag(Arow1 + kc + base0);
#pragma unroll
      for (int t = 0; t < 8; ++t) {
        const int p = cbase + t * 16 + l15;                  // score column
        const v16bf b = load_b_frag(dpT + (size_t)p * NN + kc + base0);
        acc[0][t] = WMMA_BF16(a0, b, acc[0][t]);
        acc[1][t] = WMMA_BF16(a1, b, acc[1][t]);
      }
    }
#pragma unroll
    for (int rt = 0; rt < 2; ++rt)
#pragma unroll
      for (int t = 0; t < 8; ++t)
#pragma unroll
        for (int r = 0; r < 8; ++r)
          S[(size_t)(rt * 16 + rowoff + r) * SPAD + cbase + t * 16 + l15] =
              (__bf16)(acc[rt][t][r] * scale);
  }
  __syncthreads();

  // ---- stage B: softmax stats over S (32 rows x 4096), 8 threads/row ------
  // Pass 1: row max.  Pass 2: S <- exp(S - max), accumulate row sum.
  // All LDS traffic in 16-byte (8 x bf16) quanta.
  const int r32 = tid >> 3, seg = tid & 7;
  __bf16* Srow = S + (size_t)r32 * SPAD + seg * 512;

  float mx = -3.4e38f;
  for (int i = 0; i < 512; i += 8) {
    BF16x8 u; u.q = *(const uint4*)(Srow + i);               // ds_load_b128
#pragma unroll
    for (int j = 0; j < 8; ++j) mx = fmaxf(mx, (float)u.e[j]);
  }
  red[tid] = mx;
  __syncthreads();
  if (seg == 0) {
    float m = red[tid];
    for (int i = 1; i < 8; ++i) m = fmaxf(m, red[tid + i]);
    rowstat[r32] = m;
  }
  __syncthreads();
  const float rm = rowstat[r32];
  float sum = 0.0f;
  for (int i = 0; i < 512; i += 8) {
    BF16x8 u; u.q = *(const uint4*)(Srow + i);
#pragma unroll
    for (int j = 0; j < 8; ++j) {
      const float e = __expf((float)u.e[j] - rm);
      u.e[j] = (__bf16)e;
      sum += e;
    }
    *(uint4*)(Srow + i) = u.q;                               // ds_store_b128
  }
  red[tid] = sum;
  __syncthreads();
  if (seg == 0) {
    float s = 0.0f;
    for (int i = 0; i < 8; ++i) s += red[tid + i];
    rowstat[r32] = 1.0f / s;                                 // folded in at end
  }
  __syncthreads();

  // ---- stage C: ctx = exp-matrix(32x4096) @ V_h(4096x32), 8-wave split ----
  const __bf16* VT = vbT + (size_t)h * DH * NN;              // [32][4096]
  v8f cacc[2][2];
  {
    const v8f z = {};
    cacc[0][0] = z; cacc[0][1] = z; cacc[1][0] = z; cacc[1][1] = z;
  }
  for (int kc = wid * 512; kc < wid * 512 + 512; kc += 32) {
    const v16bf b0 = load_b_frag(VT + (size_t)(l15)      * NN + kc + base0);
    const v16bf b1 = load_b_frag(VT + (size_t)(16 + l15) * NN + kc + base0);
#pragma unroll
    for (int rt = 0; rt < 2; ++rt) {
      const __bf16* Sp = S + (size_t)(rt * 16 + l15) * SPAD + kc + base0;
      BF16x16 aa;
      aa.q[0] = *(const uint4*)Sp;
      aa.q[1] = *(const uint4*)(Sp + 16);
      cacc[rt][0] = WMMA_BF16(aa.v, b0, cacc[rt][0]);
      cacc[rt][1] = WMMA_BF16(aa.v, b1, cacc[rt][1]);
    }
  }
#pragma unroll
  for (int rt = 0; rt < 2; ++rt)
#pragma unroll
    for (int r = 0; r < 8; ++r) {
      atomicAdd(&ctx[(rt * 16 + rowoff + r) * DH + l15],      cacc[rt][0][r]);
      atomicAdd(&ctx[(rt * 16 + rowoff + r) * DH + 16 + l15], cacc[rt][1][r]);
    }
  __syncthreads();

  // ---- write out [n, h*32 + j] = ctx * (1/rowsum)  (f32), 4 per thread ----
#pragma unroll
  for (int s = 0; s < 4; ++s) {
    const int i = tid + s * 256;
    const int r = i >> 5;                                    // local row
    out[(size_t)(n0 + r) * DM + h * DH + (i & 31)] = ctx[i] * rowstat[r];
  }
}

// ---------------------------------------------------------------------------
extern "C" void kernel_launch(void* const* d_in, const int* in_sizes, int n_in,
                              void* d_out, int out_size, void* d_ws, size_t ws_size,
                              hipStream_t stream) {
  const float* XQ   = (const float*)d_in[0];
  const float* XK   = (const float*)d_in[1];
  const float* XV   = (const float*)d_in[2];
  // d_in[3] = adj_matrix: dead code in the reference.
  const float* dist = (const float*)d_in[4];
  const float* WQ   = (const float*)d_in[5];
  const float* WK   = (const float*)d_in[6];
  const float* WV   = (const float*)d_in[7];
  float* out = (float*)d_out;

  char* ws = (char*)d_ws;
  const size_t szQKV = (size_t)NH * NN * DH * sizeof(__bf16);   // 2 MiB each
  const size_t szMat = (size_t)NN * NN * sizeof(__bf16);        // 32 MiB each
  __bf16* qb  = (__bf16*)(ws);
  __bf16* kb  = (__bf16*)(ws + szQKV);
  __bf16* vbT = (__bf16*)(ws + 2 * szQKV);
  __bf16* dpT = (__bf16*)(ws + 3 * szQKV);
  __bf16* ab  = (__bf16*)(ws + 3 * szQKV + szMat);
  (void)ws_size; (void)in_sizes; (void)n_in; (void)out_size;

  proj_one<<<dim3(32, 16), 256, 0, stream>>>(XQ, WQ, qb, 0);
  proj_one<<<dim3(32, 16), 256, 0, stream>>>(XK, WK, kb, 0);
  proj_one<<<dim3(32, 16), 256, 0, stream>>>(XV, WV, vbT, 1);
  rescale_t<<<dim3(128, 128), dim3(32, 8), 0, stream>>>(dist, dpT);

  const size_t smem = (size_t)ROWS * SPAD * sizeof(__bf16)   // score tile
                    + (size_t)ROWS * DH * sizeof(float)      // ctx
                    + 256 * sizeof(float)                    // red
                    + ROWS * sizeof(float);                  // rowstat
  for (int h = 0; h < NH; ++h) {
    qk_relu<<<512, 256, 0, stream>>>(qb, kb, ab, h);
    attn_ctx<<<128, 256, smem, stream>>>(ab, dpT, vbT, out, h);
  }
}
